// CausalSelfAttention_60438779789431
// MI455X (gfx1250) — compile-verified, matrix operand order checked by reading, not run on hardware
//
#include <hip/hip_runtime.h>

typedef __attribute__((ext_vector_type(16))) _Float16 v16h;
typedef __attribute__((ext_vector_type(8)))  float    v8f;

#define DEVINL __device__ __forceinline__

// ---- optional CDNA5 async global->LDS DMA path (ASYNCcnt) ----
#ifndef __has_builtin
#define __has_builtin(x) 0
#endif
#if __has_builtin(__builtin_amdgcn_global_load_async_to_lds_b128) && \
    __has_builtin(__builtin_amdgcn_s_wait_asynccnt)
#define USE_ASYNC_LDS 1
typedef int async_v4i __attribute__((vector_size(16)));
typedef __attribute__((address_space(1))) async_v4i* gv4i_p;  // global (__device__)
typedef __attribute__((address_space(3))) async_v4i* lv4i_p;  // LDS (__shared__)
#define AS1V(p) ((gv4i_p)(p))
#define AS3V(p) ((lv4i_p)(p))
#else
#define USE_ASYNC_LDS 0
#endif

DEVINL void wait_async_lds() {
#if USE_ASYNC_LDS
    __builtin_amdgcn_s_wait_asynccnt(0);
#endif
}

DEVINL unsigned short f32_to_f16u(float f) {
    _Float16 h = (_Float16)f;
    return __builtin_bit_cast(unsigned short, h);
}

union Frag16 {
    uint4 u[2];
    v16h  v;
};

// Load a 16x32 (MxK) 16-bit fragment in the CDNA5 WMMA A/B register layout:
// lane L (0..15): row M=L, K = 0..7 in VGPR0-3, K = 16..23 in VGPR4-7
// lane L+16:      row M=L, K = 8..15 and 24..31
// 'base' points to a row-major [16][stride] tile (stride in elements, 16B aligned).
DEVINL v16h load_frag(const unsigned short* base, int stride, int koff, int lane) {
    const int m    = lane & 15;
    const int half = lane >> 4;
    const unsigned short* p = base + m * stride + koff + half * 8;
    Frag16 f;
    f.u[0] = *(const uint4*)(p);        // K = half*8 .. half*8+7
    f.u[1] = *(const uint4*)(p + 16);   // K = 16+half*8 .. 16+half*8+7
    return f.v;
}

DEVINL v8f vzero8() {
    v8f z = {0.f, 0.f, 0.f, 0.f, 0.f, 0.f, 0.f, 0.f};
    return z;
}

// ---------------------------------------------------------------------------
// Elementwise f32 -> f16 conversion
// ---------------------------------------------------------------------------
__global__ void cvt_f32_to_f16_kernel(const float* __restrict__ in,
                                      unsigned short* __restrict__ out, int n) {
    int i = blockIdx.x * blockDim.x + threadIdx.x;
    int stride = gridDim.x * blockDim.x;
    for (; i < n; i += stride) out[i] = f32_to_f16u(in[i]);
}

// ---------------------------------------------------------------------------
// f16 GEMM, C = A[M,K] * B[K,N], f32 accumulate via v_wmma_f32_16x16x32_f16.
// Block tile 128x256, BK=32, 256 threads = 8 waves (2x4 wave grid), each wave
// owns a 64x64 sub-tile = 4x4 WMMA tiles (16 accumulators, 16 WMMA / K-step).
// Double-buffered LDS; A tile staged via async DMA when available.
// EPI==0: write f32 row-major to outF.
// EPI==1: scatter f16 into Q/K/V buffers laid out [B,H,T,64] (QKV projection).
// ---------------------------------------------------------------------------
template <int EPI>
__global__ __launch_bounds__(256)
void gemm_f16_kernel(const unsigned short* __restrict__ A,
                     const unsigned short* __restrict__ Bm,
                     float* __restrict__ outF,
                     unsigned short* __restrict__ outQ,
                     unsigned short* __restrict__ outK,
                     unsigned short* __restrict__ outV,
                     int M, int N, int K) {
    __shared__ __align__(16) unsigned short Alds[2][128 * 32];  // [m][k]
    __shared__ __align__(16) unsigned short Blds[2][256 * 32];  // [n][k] transposed

    const int tid  = threadIdx.x;
    const int lane = tid & 31;
    const int wave = tid >> 5;
    const int m0   = blockIdx.y * 128;
    const int n0   = blockIdx.x * 256;
    const int wm   = (wave & 1) * 64;   // wave row offset within block tile
    const int wn   = (wave >> 1) * 64;  // wave col offset within block tile

    v8f acc[4][4];
#pragma unroll
    for (int i = 0; i < 4; ++i)
#pragma unroll
        for (int j = 0; j < 4; ++j) acc[i][j] = vzero8();

    auto stage = [&](int k0, int buf) {
        // ---- A tile [128][32] row-major: straight copy ----
        {
            const int r = tid >> 1;        // 0..127
            const int c = (tid & 1) * 16;  // 0 or 16
            const unsigned short* g = A + (size_t)(m0 + r) * K + k0 + c;
            unsigned short* l = &Alds[buf][r * 32 + c];
#if USE_ASYNC_LDS
            __builtin_amdgcn_global_load_async_to_lds_b128(AS1V(g), AS3V(l), 0, 0);
            __builtin_amdgcn_global_load_async_to_lds_b128(AS1V(g + 8), AS3V(l + 8), 0, 0);
#else
            const uint4* src = (const uint4*)g;
            uint4 a0 = src[0];
            uint4 a1 = src[1];
            uint4* dst = (uint4*)l;
            dst[0] = a0;
            dst[1] = a1;
#endif
        }
        // ---- B tile transposed into Blds[n][k] (must go through VGPRs) ----
        {
            const int nn = (tid & 31) * 8;  // 0..248
            const int kk = tid >> 5;        // 0..7
#pragma unroll
            for (int s = 0; s < 4; ++s) {
                const int kg = k0 + kk + s * 8;
                uint4 bvec = *(const uint4*)(Bm + (size_t)kg * N + n0 + nn);
                const unsigned short* bs = (const unsigned short*)&bvec;
#pragma unroll
                for (int i = 0; i < 8; ++i)
                    Blds[buf][(nn + i) * 32 + kk + s * 8] = bs[i];
            }
        }
    };

    const int nk = K >> 5;
    stage(0, 0);
    wait_async_lds();
    __syncthreads();

    for (int kt = 0; kt < nk; ++kt) {
        const int buf = kt & 1;
        if (kt + 1 < nk) stage((kt + 1) * 32, buf ^ 1);  // overlap DMA with math

        v16h af[4], bfr[4];
#pragma unroll
        for (int i = 0; i < 4; ++i)
            af[i] = load_frag(&Alds[buf][(wm + i * 16) * 32], 32, 0, lane);
#pragma unroll
        for (int j = 0; j < 4; ++j)
            bfr[j] = load_frag(&Blds[buf][(wn + j * 16) * 32], 32, 0, lane);

#pragma unroll
        for (int i = 0; i < 4; ++i)
#pragma unroll
            for (int j = 0; j < 4; ++j)
                acc[i][j] = __builtin_amdgcn_wmma_f32_16x16x32_f16(
                    false, af[i], false, bfr[j], (short)0, acc[i][j], false, false);

        wait_async_lds();
        __syncthreads();
    }

    // ---- epilogue ----
    const int half = lane >> 4;
    const int nl   = lane & 15;
#pragma unroll
    for (int i = 0; i < 4; ++i) {
#pragma unroll
        for (int j = 0; j < 4; ++j) {
#pragma unroll
            for (int r = 0; r < 8; ++r) {
                const int mg = m0 + wm + i * 16 + half * 8 + r;
                const int ng = n0 + wn + j * 16 + nl;
                const float v = acc[i][j][r];
                if (EPI == 0) {
                    outF[(size_t)mg * N + ng] = v;
                } else {
                    // ng in [0, 3072): segment 0=Q 1=K 2=V, layout [B,H,T,64]
                    const int seg  = ng >> 10;
                    const int dcol = ng & 1023;
                    const int h    = dcol >> 6;
                    const int hd   = dcol & 63;
                    const int b    = mg >> 11;
                    const int t    = mg & 2047;
                    const size_t idx =
                        (((size_t)(b * 16 + h) * 2048) + t) * 64 + hd;
                    unsigned short hv = f32_to_f16u(v);
                    (seg == 0 ? outQ : (seg == 1 ? outK : outV))[idx] = hv;
                }
            }
        }
    }
}

// ---------------------------------------------------------------------------
// Flash attention: block = 128 query rows of one (b,h); 8 waves, each wave
// owns a 16-row query strip. Q fragments persist in registers; K/V tiles of
// 32 keys staged in LDS (K via async DMA, V transposed through VGPRs);
// online softmax; P restaged via per-wave LDS scratch.
// ---------------------------------------------------------------------------
__global__ __launch_bounds__(256)
void flash_attn_kernel(const unsigned short* __restrict__ Qb,
                       const unsigned short* __restrict__ Kb,
                       const unsigned short* __restrict__ Vb,
                       unsigned short* __restrict__ Yb) {
    __shared__ __align__(16) unsigned short Klds[32 * 64];      // [key][hd]
    __shared__ __align__(16) unsigned short Vt[64 * 32];        // [hd][key]
    __shared__ __align__(16) unsigned short Plds[8 * 16 * 32];  // per-wave [16][32]

    const int tid  = threadIdx.x;
    const int lane = tid & 31;
    const int wave = tid >> 5;
    const int bh   = blockIdx.y;          // b*16 + h
    const int q0b  = blockIdx.x * 128;
    const int t0   = q0b + wave * 16;     // this wave's query strip start
    const size_t headBase = (size_t)bh * 2048 * 64;

    const int half = lane >> 4;
    const int nl   = lane & 15;

    // persistent Q fragments: 16 rows x 64 (two K-steps of 32)
    v16h qf[2];
#pragma unroll
    for (int s = 0; s < 2; ++s)
        qf[s] = load_frag(Qb + headBase + (size_t)t0 * 64, 64, s * 32, lane);

    v8f o[4];
#pragma unroll
    for (int j = 0; j < 4; ++j) o[j] = vzero8();
    float Mrow[8], Lrow[8];
#pragma unroll
    for (int r = 0; r < 8; ++r) { Mrow[r] = -1e30f; Lrow[r] = 0.f; }

    const int nkt = (q0b + 128) >> 5;  // 32-key tiles covering block's causal range

    for (int kt = 0; kt < nkt; ++kt) {
        const int kbase = kt * 32;
        // ---- cooperative K / V^T staging ----
        {
            const int r = tid >> 3;  // key row 0..31
            const int c = tid & 7;   // 8-elem chunk 0..7
            const size_t gofs = headBase + (size_t)(kbase + r) * 64 + c * 8;
#if USE_ASYNC_LDS
            __builtin_amdgcn_global_load_async_to_lds_b128(
                AS1V(Kb + gofs), AS3V(Klds + r * 64 + c * 8), 0, 0);
#else
            *(uint4*)(Klds + r * 64 + c * 8) = *(const uint4*)(Kb + gofs);
#endif
            uint4 vv = *(const uint4*)(Vb + gofs);
            const unsigned short* vs = (const unsigned short*)&vv;
#pragma unroll
            for (int i = 0; i < 8; ++i)
                Vt[(c * 8 + i) * 32 + r] = vs[i];
        }
        wait_async_lds();
        __syncthreads();

        if (kbase <= t0 + 15) {  // wave-uniform causal activity test
            // ---- S = Q * K^T : two 16-col tiles, K-dim 64 in two steps ----
            v8f s0 = vzero8(), s1 = vzero8();
#pragma unroll
            for (int s = 0; s < 2; ++s) {
                v16h k0f = load_frag(Klds + 0 * 64, 64, s * 32, lane);
                v16h k1f = load_frag(Klds + 16 * 64, 64, s * 32, lane);
                s0 = __builtin_amdgcn_wmma_f32_16x16x32_f16(
                    false, qf[s], false, k0f, (short)0, s0, false, false);
                s1 = __builtin_amdgcn_wmma_f32_16x16x32_f16(
                    false, qf[s], false, k1f, (short)0, s1, false, false);
            }
            // ---- scale + causal mask ----
            const float scale = 0.125f;  // 1/sqrt(64)
            const bool diag = (kbase + 31 > t0);
#pragma unroll
            for (int r = 0; r < 8; ++r) {
                const int mg = t0 + half * 8 + r;
                float a  = s0[r] * scale;
                float b2 = s1[r] * scale;
                if (diag) {
                    if (kbase + nl > mg)      a  = -1e30f;
                    if (kbase + 16 + nl > mg) b2 = -1e30f;
                }
                s0[r] = a;
                s1[r] = b2;
            }
            // ---- online softmax (row reductions across 16-lane groups) ----
            unsigned short* pw = Plds + wave * (16 * 32);
#pragma unroll
            for (int r = 0; r < 8; ++r) {
                float v = fmaxf(s0[r], s1[r]);
#pragma unroll
                for (int off = 1; off < 16; off <<= 1)
                    v = fmaxf(v, __shfl_xor(v, off, 16));
                const float mnew  = fmaxf(Mrow[r], v);
                const float alpha = __expf(Mrow[r] - mnew);
                const float p0    = __expf(s0[r] - mnew);
                const float p1    = __expf(s1[r] - mnew);
                float sum = p0 + p1;
#pragma unroll
                for (int off = 1; off < 16; off <<= 1)
                    sum += __shfl_xor(sum, off, 16);
                Lrow[r] = Lrow[r] * alpha + sum;
                Mrow[r] = mnew;
#pragma unroll
                for (int j = 0; j < 4; ++j) o[j][r] *= alpha;
                const int m = half * 8 + r;
                pw[m * 32 + nl]      = f32_to_f16u(p0);
                pw[m * 32 + 16 + nl] = f32_to_f16u(p1);
            }
            // wave-private LDS store->load: drain DS counter before reload
            asm volatile("s_wait_dscnt 0" ::: "memory");
            v16h pf = load_frag(pw, 32, 0, lane);
            // ---- O += P * V ----
#pragma unroll
            for (int j = 0; j < 4; ++j) {
                v16h vf = load_frag(Vt + (j * 16) * 32, 32, 0, lane);
                o[j] = __builtin_amdgcn_wmma_f32_16x16x32_f16(
                    false, pf, false, vf, (short)0, o[j], false, false);
            }
        }
        __syncthreads();
    }

    // ---- normalize and store Y as f16 [B,T,D] (heads re-interleaved) ----
    const int b = bh >> 4;
    const int h = bh & 15;
#pragma unroll
    for (int j = 0; j < 4; ++j) {
#pragma unroll
        for (int r = 0; r < 8; ++r) {
            const int m = half * 8 + r;
            const float v = o[j][r] / Lrow[r];
            const size_t idx =
                ((size_t)(b * 2048 + t0 + m)) * 1024 + h * 64 + j * 16 + nl;
            Yb[idx] = f32_to_f16u(v);
        }
    }
}

// ---------------------------------------------------------------------------
// Host launcher
// ---------------------------------------------------------------------------
extern "C" void kernel_launch(void* const* d_in, const int* in_sizes, int n_in,
                              void* d_out, int out_size, void* d_ws, size_t ws_size,
                              hipStream_t stream) {
    (void)in_sizes; (void)n_in; (void)out_size; (void)ws_size;
    const float* x     = (const float*)d_in[0];   // [4,2048,1024]
    const float* w_qkv = (const float*)d_in[1];   // [1024,3072]
    const float* w_out = (const float*)d_in[2];   // [1024,1024]
    float* out = (float*)d_out;                   // [4,2048,1024]

    const int BT = 4 * 2048;   // 8192 tokens
    const int D  = 1024;

    char* w = (char*)d_ws;
    auto take = [&](size_t bytes) {
        void* p = (void*)w;
        w += (bytes + 255) & ~(size_t)255;
        return p;
    };
    unsigned short* xb    = (unsigned short*)take((size_t)BT * D * 2);
    unsigned short* wqkvb = (unsigned short*)take((size_t)D * 3 * D * 2);
    unsigned short* woutb = (unsigned short*)take((size_t)D * D * 2);
    unsigned short* Qb    = (unsigned short*)take((size_t)BT * D * 2);
    unsigned short* Kb    = (unsigned short*)take((size_t)BT * D * 2);
    unsigned short* Vb    = (unsigned short*)take((size_t)BT * D * 2);
    unsigned short* Yb    = (unsigned short*)take((size_t)BT * D * 2);

    cvt_f32_to_f16_kernel<<<1024, 256, 0, stream>>>(x, xb, BT * D);
    cvt_f32_to_f16_kernel<<<1024, 256, 0, stream>>>(w_qkv, wqkvb, D * 3 * D);
    cvt_f32_to_f16_kernel<<<512, 256, 0, stream>>>(w_out, woutb, D * D);

    // QKV projection: [8192,1024] x [1024,3072] -> scatter Q/K/V [B,H,T,64] f16
    gemm_f16_kernel<1><<<dim3(3072 / 256, BT / 128), 256, 0, stream>>>(
        xb, wqkvb, nullptr, Qb, Kb, Vb, BT, 3 * D, D);

    // Flash attention: grid = (T/128 query blocks, B*H heads)
    flash_attn_kernel<<<dim3(2048 / 128, 4 * 16), 256, 0, stream>>>(Qb, Kb, Vb, Yb);

    // Output projection: [8192,1024] x [1024,1024] -> f32 d_out
    gemm_f16_kernel<0><<<dim3(D / 256, BT / 128), 256, 0, stream>>>(
        Yb, woutb, out, nullptr, nullptr, nullptr, BT, D, D);
}
